// CausalVAE_11733850652839
// MI455X (gfx1250) — matrix operand order repaired and checked by reading, not compiled, not run on hardware
//
#include <hip/hip_runtime.h>
#include <hip/hip_bf16.h>

// ---------------------------------------------------------------------------
// CausalVAE fused kernel for MI455X (gfx1250, wave32).
// v2: Tensor Data Mover stages W1[i]/W2[i] into LDS (once per block instead of
// once per wave -> 8x less L2 traffic), with TDM pad-on-load giving a 528B row
// stride so B-fragment ds_load_b128s are LDS-bank-conflict-free. W2 is split
// into two N-halves and ping-pong double-buffered so TDM copy overlaps WMMA.
// Math stays on v_wmma_f32_16x16x32_bf16 with f32 accumulate.
// ---------------------------------------------------------------------------

typedef __attribute__((ext_vector_type(16))) __bf16 v16bf;
typedef __attribute__((ext_vector_type(8)))  __bf16 v8bf;
typedef __attribute__((ext_vector_type(8)))  float  v8f;
typedef __attribute__((ext_vector_type(4)))  unsigned int u32x4;
typedef __attribute__((ext_vector_type(8)))  int i32x8;
typedef __attribute__((ext_vector_type(4)))  int i32x4;

#define DD     32
#define DF     256
#define BATCH  32768
#define TILE_M 128          // rows per block (8 waves x 16 rows)
#define RS     264          // LDS activation stride (bf16 elems), padded
#define YS     36           // LDS y-buffer stride (f32), padded
#define ZS     68           // LDS z-buffer stride (f32), padded
#define W1S    40           // staged W1 stride (32 + 8 pad bf16) = 80B
#define W2S    264          // staged W2 stride (256 + 8 pad bf16) = 528B

// LDS memory map (bytes)
#define OFF_ACT 0                       // 128*264*2 = 67584
#define OFF_Y   67584                   // 128*36*4  = 18432
#define OFF_W3S 86016                   // 256*4     = 1024
#define OFF_W1B 87040                   // 256*40*2  = 20480
#define OFF_W2A 107520                  // 128*264*2 = 67584
#define OFF_W2B 175104                  // 128*264*2 = 67584
#define SMEM_BYTES 242688

// workspace layout (bf16 elements): transposed bf16 weights Wt[N][K]
#define OFF_WE1T 0
#define OFF_WE2T (OFF_WE1T + 256*32)
#define OFF_WE3T (OFF_WE2T + 256*256)
#define OFF_W1T  (OFF_WE3T + 64*256)
#define OFF_W2T  (OFF_W1T  + 32*256*32)

#if defined(__has_builtin)
#  if __has_builtin(__builtin_amdgcn_tensor_load_to_lds)
#    define HAS_TDM 1
#  endif
#endif
#ifndef HAS_TDM
#  define HAS_TDM 0
#endif

#if HAS_TDM
#define TDM_WAIT(N) do { if (wave == 0) __builtin_amdgcn_s_wait_tensorcnt((short)(N)); } while (0)
#else
#define TDM_WAIT(N) do { } while (0)
#endif

// ---- prep: convert f32 [B][K][N] -> bf16 transposed [B][N][K] --------------
__global__ void transpose_to_bf16(const float* __restrict__ in,
                                  __bf16* __restrict__ out,
                                  int B, int K, int N) {
  size_t idx = (size_t)blockIdx.x * blockDim.x + threadIdx.x;
  size_t total = (size_t)B * K * N;
  if (idx >= total) return;
  int kn = K * N;
  int b = (int)(idx / kn);
  int r = (int)(idx % kn);
  int n = r / K;
  int k = r % K;
  out[idx] = (__bf16)in[(size_t)b * kn + (size_t)k * N + n];
}

// ---- TDM: 2D tile global->LDS with pad-on-load (D# per ISA ch.8) -----------
#if HAS_TDM
__device__ __forceinline__ void tdm_load_2d(const __bf16* src, void* lds_dst,
                                            unsigned k_elems, unsigned n_rows,
                                            unsigned pad_interval, unsigned pad_amount) {
  unsigned long long ga = (unsigned long long)(const void*)src;
  unsigned lds = (unsigned)(unsigned long long)lds_dst;   // low 32b of flat = LDS offset
  u32x4 g0;
  g0[0] = 1u;                                             // count=1, user descriptor
  g0[1] = lds;                                            // lds_addr
  g0[2] = (unsigned)ga;                                   // global_addr[31:0]
  g0[3] = ((unsigned)(ga >> 32) & 0x01FFFFFFu) | (2u << 30); // addr[56:32] | type=2
  i32x8 g1;
  g1[0] = (int)((1u << 16) |                              // data_size = 2 bytes (bf16)
                (1u << 20) |                              // pad_enable
                (pad_interval << 22) | (pad_amount << 25));
  g1[1] = (int)((k_elems & 0xFFFFu) << 16);               // tensor_dim0[15:0] @ [63:48]
  g1[2] = (int)(((k_elems >> 16) & 0xFFFFu) |             // tensor_dim0[31:16]
                ((n_rows & 0xFFFFu) << 16));              // tensor_dim1[15:0]
  g1[3] = (int)(((n_rows >> 16) & 0xFFFFu) |              // tensor_dim1[31:16]
                ((k_elems & 0xFFFFu) << 16));             // tile_dim0 = k_elems
  g1[4] = (int)(n_rows & 0xFFFFu);                        // tile_dim1 = n_rows, tile_dim2=0
  g1[5] = (int)k_elems;                                   // tensor_dim0_stride[31:0]
  g1[6] = 0;
  g1[7] = 0;
  i32x4 z4 = {0, 0, 0, 0};
#if __clang_major__ >= 23
  i32x8 z8 = {0, 0, 0, 0, 0, 0, 0, 0};
  __builtin_amdgcn_tensor_load_to_lds(g0, g1, z4, z4, z8, 0);
#else
  __builtin_amdgcn_tensor_load_to_lds(g0, g1, z4, z4, 0);
#endif
}
#endif

__device__ __forceinline__ void stage_weights(const __bf16* src, __bf16* dst,
                                              unsigned k, unsigned n, int stride,
                                              unsigned pi, unsigned pa,
                                              int wave, int tid) {
#if HAS_TDM
  (void)stride; (void)tid;
  if (wave == 0) tdm_load_2d(src, dst, k, n, pi, pa);
#else
  (void)wave; (void)pi; (void)pa;
  int total = (int)(k * n);
  for (int t = tid; t < total; t += 256) {
    int r = t / (int)k, c = t % (int)k;
    dst[(size_t)r * stride + c] = src[t];
  }
#endif
}

// ---- fragment helpers ------------------------------------------------------
// A layout (16-bit, 16x32): lane l15 = row; lanes 0-15 hold K [k0..k0+7] then
// [k0+16..k0+23]; lanes 16-31 hold [k0+8..15] then [k0+24..31].
template <int KSTEPS>
__device__ __forceinline__ void load_A(const __bf16* act, int stride,
                                       int m0, int l15, int hi, v16bf* A) {
  const __bf16* base = act + (size_t)(m0 + l15) * stride + hi * 8;
#pragma unroll
  for (int kt = 0; kt < KSTEPS; ++kt) {
    v8bf a0 = *(const v8bf*)(base + kt * 32);
    v8bf a1 = *(const v8bf*)(base + kt * 32 + 16);
    A[kt] = __builtin_shufflevector(a0, a1, 0,1,2,3,4,5,6,7,8,9,10,11,12,13,14,15);
  }
}

// B layout (16-bit, 32x16): lane l15 = column; lanes 0-15 K=[k0..k0+7] then
// [k0+16..k0+23]; lanes 16-31 K=[k0+8..15] then [k0+24..31] (matches A).
template <int KSTEPS, int NTILES>
__device__ __forceinline__ void gemm_global(const v16bf* A, const __bf16* Wt,
                                            int Kdim, int l15, int hi, v8f* C) {
#pragma unroll
  for (int nt = 0; nt < NTILES; ++nt) {
    v8f c = {0.f,0.f,0.f,0.f,0.f,0.f,0.f,0.f};
    const __bf16* colp = Wt + (size_t)(nt * 16 + l15) * Kdim + hi * 8;
#pragma unroll
    for (int kt = 0; kt < KSTEPS; ++kt) {
      v8bf b0 = *(const v8bf*)(colp + kt * 32);
      v8bf b1 = *(const v8bf*)(colp + kt * 32 + 16);
      v16bf b = __builtin_shufflevector(b0, b1, 0,1,2,3,4,5,6,7,8,9,10,11,12,13,14,15);
      c = __builtin_amdgcn_wmma_f32_16x16x32_bf16(false, A[kt], false, b,
                                                  (short)0, c, false, false);
    }
    C[nt] = c;
  }
}

template <int KSTEPS, int NTILES>
__device__ __forceinline__ void gemm_lds(const v16bf* A, const __bf16* Wl,
                                         int stride, int l15, int hi, v8f* C) {
#pragma unroll
  for (int nt = 0; nt < NTILES; ++nt) {
    v8f c = {0.f,0.f,0.f,0.f,0.f,0.f,0.f,0.f};
    const __bf16* colp = Wl + (size_t)(nt * 16 + l15) * stride + hi * 8;
#pragma unroll
    for (int kt = 0; kt < KSTEPS; ++kt) {
      v8bf b0 = *(const v8bf*)(colp + kt * 32);
      v8bf b1 = *(const v8bf*)(colp + kt * 32 + 16);
      v16bf b = __builtin_shufflevector(b0, b1, 0,1,2,3,4,5,6,7,8,9,10,11,12,13,14,15);
      c = __builtin_amdgcn_wmma_f32_16x16x32_bf16(false, A[kt], false, b,
                                                  (short)0, c, false, false);
    }
    C[nt] = c;
  }
}

// C/D layout: VGPR r -> row m0 + hi*8 + r, column nBase + nt*16 + l15.
template <int NTILES, bool RELU>
__device__ __forceinline__ void store_act(const v8f* C, const float* bias, int nBase,
                                          __bf16* act, int stride,
                                          int m0, int l15, int hi) {
#pragma unroll
  for (int nt = 0; nt < NTILES; ++nt) {
    float bv = bias[nBase + nt * 16 + l15];
#pragma unroll
    for (int r = 0; r < 8; ++r) {
      float v = C[nt][r] + bv;
      if (RELU) v = v > 0.f ? v : 0.f;
      act[(size_t)(m0 + hi * 8 + r) * stride + nBase + nt * 16 + l15] = (__bf16)v;
    }
  }
}

template <int NTILES>
__device__ __forceinline__ void store_f32(const v8f* C, const float* bias,
                                          float* buf, int stride,
                                          int m0, int l15, int hi) {
#pragma unroll
  for (int nt = 0; nt < NTILES; ++nt) {
    float bv = bias[nt * 16 + l15];
#pragma unroll
    for (int r = 0; r < 8; ++r)
      buf[(size_t)(m0 + hi * 8 + r) * stride + nt * 16 + l15] = C[nt][r] + bv;
  }
}

// ---- fused encoder + reparameterize + 32-step causal decoder ---------------
__global__ __launch_bounds__(256) void causal_vae_fused(
    const float* __restrict__ X,   const float* __restrict__ EPS,
    const float* __restrict__ be1, const float* __restrict__ be2,
    const float* __restrict__ be3, const float* __restrict__ b1,
    const float* __restrict__ b2,  const float* __restrict__ W3,
    const float* __restrict__ b3,  const __bf16* __restrict__ wsb,
    float* __restrict__ out) {
  extern __shared__ char smem[];
  __bf16* actbuf = (__bf16*)(smem + OFF_ACT);
  float*  ybuf   = (float*)(smem + OFF_Y);
  float*  w3s    = (float*)(smem + OFF_W3S);
  __bf16* w1buf  = (__bf16*)(smem + OFF_W1B);
  __bf16* w2A    = (__bf16*)(smem + OFF_W2A);
  __bf16* w2B    = (__bf16*)(smem + OFF_W2B);
  float*  zbuf   = (float*)(smem + OFF_W1B);   // encoder-only, overlays W1/W2A

  const int tid  = threadIdx.x;
  const int wave = tid >> 5;
  const int lane = tid & 31;
  const int l15  = lane & 15;
  const int hi   = lane >> 4;
  const int m0   = wave * 16;
  const int rowbase = blockIdx.x * TILE_M;

  // ---- Encoder L1: h1 = relu(x @ We1 + be1), K=32 ----
  {
    v16bf A1;
    const float* p = X + (size_t)(rowbase + m0 + l15) * DD + hi * 8;
#pragma unroll
    for (int j = 0; j < 8; ++j) { A1[j] = (__bf16)p[j]; A1[8 + j] = (__bf16)p[16 + j]; }
    v8f C[16];
    gemm_global<1, 16>(&A1, wsb + OFF_WE1T, 32, l15, hi, C);
    store_act<16, true>(C, be1, 0, actbuf, RS, m0, l15, hi);
  }
  // ---- Encoder L2: h2 = relu(h1 @ We2 + be2), K=256 ----
  {
    v16bf A[8];
    load_A<8>(actbuf, RS, m0, l15, hi, A);
    v8f C[16];
    gemm_global<8, 16>(A, wsb + OFF_WE2T, 256, l15, hi, C);
    store_act<16, true>(C, be2, 0, actbuf, RS, m0, l15, hi);
  }
  // ---- Encoder L3: z = h2 @ We3 + be3 (N=64, no relu) ----
  {
    v16bf A[8];
    load_A<8>(actbuf, RS, m0, l15, hi, A);
    v8f C[4];
    gemm_global<8, 4>(A, wsb + OFF_WE3T, 256, l15, hi, C);
    store_f32<4>(C, be3, zbuf, ZS, m0, l15, hi);
  }
  // ---- Reparameterize: y0 = mu + exp(lv/2)*eps; emit mu, logvar ----
  {
    int row = m0 + l15;
    size_t grow = (size_t)rowbase + row;
#pragma unroll
    for (int c = 0; c < 16; ++c) {
      int col = hi * 16 + c;
      float mu = zbuf[(size_t)row * ZS + col];
      float lv = zbuf[(size_t)row * ZS + 32 + col];
      float e  = EPS[grow * DD + col];
      out[(size_t)1 * BATCH * DD + grow * DD + col] = mu;
      out[(size_t)2 * BATCH * DD + grow * DD + col] = lv;
      ybuf[(size_t)row * YS + col] = mu + __expf(lv * 0.5f) * e;
    }
  }
  __syncthreads();   // zbuf dead; decoder weight buffers may now be written

  // ---- Decoder prologue: TDM-stage W1[0] and W2[0] half0 ----
  stage_weights(wsb + OFF_W1T, w1buf, 32, 256, W1S, /*pi=*/3, /*pa=*/3, wave, tid);
  stage_weights(wsb + OFF_W2T, w2A, 256, 128, W2S, /*pi=*/6, /*pa=*/3, wave, tid);

  // ---- Causal decoder: 32 sequential node updates ----
  for (int i = 0; i < DD; ++i) {
    TDM_WAIT(1);            // W1[i] staged (TDM completes in issue order)
    __syncthreads();

    // D1: t1 = relu((y * mask<=i) @ W1[i] + b1[i]), K=32  (B from LDS)
    {
      v16bf Ay;
      const float* py = ybuf + (size_t)(m0 + l15) * YS;
#pragma unroll
      for (int j = 0; j < 8; ++j) {
        int c0 = hi * 8 + j;
        int c1 = 16 + hi * 8 + j;
        Ay[j]     = (__bf16)(c0 <= i ? py[c0] : 0.f);
        Ay[8 + j] = (__bf16)(c1 <= i ? py[c1] : 0.f);
      }
      v8f C[16];
      gemm_lds<1, 16>(&Ay, w1buf, W1S, l15, hi, C);
      store_act<16, true>(C, b1 + (size_t)i * DF, 0, actbuf, RS, m0, l15, hi);
    }

    TDM_WAIT(0);            // W2[i] half0 staged
    __syncthreads();
    // prefetch W2[i] half1 into the other buffer (overlaps D2a compute)
    stage_weights(wsb + OFF_W2T + (size_t)i * 256 * 256 + 128 * 256,
                  w2B, 256, 128, W2S, 6, 3, wave, tid);
#if !HAS_TDM
    __syncthreads();
#endif

    // D2: t2 = relu(t1 @ W2[i] + b2[i]), K=256 (dominant cost; B from LDS)
    v16bf A[8];
    load_A<8>(actbuf, RS, m0, l15, hi, A);
    {
      v8f C[8];
      gemm_lds<8, 8>(A, w2A, W2S, l15, hi, C);          // output cols 0..127
      store_act<8, true>(C, b2 + (size_t)i * DF, 0, actbuf, RS, m0, l15, hi);
    }
    __syncthreads();        // all waves done reading w2A
    TDM_WAIT(0);            // half1 staged
    if (i + 1 < DD) {       // prefetch next iteration (overlaps D2b + D3)
      stage_weights(wsb + OFF_W1T + (size_t)(i + 1) * 256 * 32,
                    w1buf, 32, 256, W1S, 3, 3, wave, tid);
      stage_weights(wsb + OFF_W2T + (size_t)(i + 1) * 256 * 256,
                    w2A, 256, 128, W2S, 6, 3, wave, tid);
    }
    __syncthreads();
    {
      v8f C[8];
      gemm_lds<8, 8>(A, w2B, W2S, l15, hi, C);          // output cols 128..255
      store_act<8, true>(C, b2 + (size_t)i * DF, 128, actbuf, RS, m0, l15, hi);
    }

    // D3: y[:, i] = t2 @ W3[i] + b3[i]
    if (tid < DF) w3s[tid] = W3[(size_t)i * DF + tid];
    __syncthreads();
    {
      int row = m0 + l15;
      const __bf16* pa = actbuf + (size_t)row * RS + hi * 128;
      const float*  pw = w3s + hi * 128;
      float s = 0.f;
#pragma unroll 8
      for (int k = 0; k < 128; ++k) s += (float)pa[k] * pw[k];
      s += __shfl_xor(s, 16);
      if (hi == 0) ybuf[(size_t)row * YS + i] = s + b3[i];
    }
    __syncthreads();
  }

  // ---- Emit y (output slots 0 and 3) ----
  {
    int row = m0 + l15;
    size_t grow = (size_t)rowbase + row;
#pragma unroll
    for (int c = 0; c < 16; ++c) {
      int col = hi * 16 + c;
      float v = ybuf[(size_t)row * YS + col];
      out[grow * DD + col] = v;
      out[(size_t)3 * BATCH * DD + grow * DD + col] = v;
    }
  }
}

extern "C" void kernel_launch(void* const* d_in, const int* in_sizes, int n_in,
                              void* d_out, int out_size, void* d_ws, size_t ws_size,
                              hipStream_t stream) {
  const float* X   = (const float*)d_in[0];
  // d_in[1] = m (unused by reference)
  const float* EPS = (const float*)d_in[2];
  const float* We1 = (const float*)d_in[3];
  const float* be1 = (const float*)d_in[4];
  const float* We2 = (const float*)d_in[5];
  const float* be2 = (const float*)d_in[6];
  const float* We3 = (const float*)d_in[7];
  const float* be3 = (const float*)d_in[8];
  const float* W1  = (const float*)d_in[9];
  const float* b1  = (const float*)d_in[10];
  const float* W2  = (const float*)d_in[11];
  const float* b2  = (const float*)d_in[12];
  const float* W3  = (const float*)d_in[13];
  const float* b3  = (const float*)d_in[14];
  float*  out = (float*)d_out;
  __bf16* wsb = (__bf16*)d_ws;

  auto launchT = [&](const float* in, size_t off, int B, int K, int N) {
    size_t total = (size_t)B * K * N;
    int blk = 256;
    int grd = (int)((total + blk - 1) / blk);
    transpose_to_bf16<<<grd, blk, 0, stream>>>(in, wsb + off, B, K, N);
  };
  launchT(We1, OFF_WE1T, 1, 32, 256);
  launchT(We2, OFF_WE2T, 1, 256, 256);
  launchT(We3, OFF_WE3T, 1, 256, 64);
  launchT(W1,  OFF_W1T, 32, 32, 256);
  launchT(W2,  OFF_W2T, 32, 256, 256);

  hipFuncSetAttribute((const void*)causal_vae_fused,
                      hipFuncAttributeMaxDynamicSharedMemorySize, SMEM_BYTES);
  causal_vae_fused<<<BATCH / TILE_M, 256, SMEM_BYTES, stream>>>(
      X, EPS, be1, be2, be3, b1, b2, W3, b3, wsb, out);
}
